// SVDNet_58205396795620
// MI455X (gfx1250) — matrix operand order repaired and checked by reading, not compiled, not run on hardware
//
#include <hip/hip_runtime.h>
#include <hip/hip_bf16.h>
#include <stdint.h>

// ---------------------------------------------------------------------------
// SVDNet for MI455X (gfx1250): complex convs collapsed to single convs and run
// as implicit-GEMM with v_wmma_f32_16x16x32_bf16 (fp32 accumulate).
// Activations are NHWC bf16 so im2col k-chunks are contiguous -> b128 staging.
// ---------------------------------------------------------------------------

#define BATCH 128
#define HH 64
#define WW 64
#define HW 4096           // HH*WW
#define NPIX (BATCH * HW) // 524288 pixels
#define RANK 8

typedef __attribute__((ext_vector_type(16))) __bf16 v16bf;
typedef __attribute__((ext_vector_type(8)))  float  v8f;

union FragBF {
  v16bf v;
  unsigned short u[16];
  unsigned int   d[8];
};

__device__ __forceinline__ unsigned short f32_to_bf16(float f) {
  unsigned int u = __float_as_uint(f);
  unsigned int lsb = (u >> 16) & 1u;
  u += 0x7fffu + lsb;                 // round-to-nearest-even
  return (unsigned short)(u >> 16);
}

// ---------------------------------------------------------------------------
// Weight pack: wcat[n][k] bf16, n in [0,2*Cout), k = (dy*3+dx)*Cin + c, padded
// to Kpad with zeros. n<Cout -> wr-wi ; n>=Cout -> wr+wi.
// ---------------------------------------------------------------------------
__global__ void svd_pack_weights(const float* __restrict__ wr,
                                 const float* __restrict__ wi,
                                 unsigned short* __restrict__ wpk,
                                 int Cin, int Cout, int Kpad) {
  int idx = blockIdx.x * blockDim.x + threadIdx.x;
  int total = 2 * Cout * Kpad;
  if (idx >= total) return;
  int n = idx / Kpad;
  int k = idx - n * Kpad;
  int K = Cin * 9;
  float v = 0.f;
  if (k < K) {
    int c = k % Cin;
    int dxy = k / Cin;                 // dy*3+dx, 0..8
    int co = (n < Cout) ? n : (n - Cout);
    float a = wr[(co * Cin + c) * 9 + dxy];
    float b = wi[(co * Cin + c) * 9 + dxy];
    v = (n < Cout) ? (a - b) : (a + b);
  }
  wpk[(size_t)n * Kpad + k] = f32_to_bf16(v);
}

__global__ void svd_pack_bias(const float* __restrict__ br,
                              const float* __restrict__ bi,
                              float* __restrict__ bcat, int Cout) {
  int n = blockIdx.x * blockDim.x + threadIdx.x;
  if (n >= 2 * Cout) return;
  int co = (n < Cout) ? n : (n - Cout);
  bcat[n] = (n < Cout) ? (br[co] - bi[co]) : (br[co] + bi[co]);
}

// x (B,H,W,2) fp32 is already NHWC -> pure cast to bf16
__global__ void svd_pack_input(const float* __restrict__ x,
                               unsigned short* __restrict__ act) {
  int idx = blockIdx.x * blockDim.x + threadIdx.x;   // B*HW*2
  if (idx >= BATCH * HW * 2) return;
  act[idx] = f32_to_bf16(x[idx]);
}

// ---------------------------------------------------------------------------
// Implicit-GEMM conv, 3x3 SAME.  M = pixels (tile 256 = four image rows),
// N = Cout2 (tile 64), K = Cin*9 padded to mult of 32.
// 256 threads = 8 wave32s in a 4(M)x2(N) grid; each wave: 64x32 tile via
// 8x v_wmma_f32_16x16x32_bf16 per 32-wide k-step.  Double-buffered LDS.
// Activations NHWC bf16: a 32-wide k-chunk (one tap, 32 channels) is
// contiguous -> b128 staging for Cin>=32; scalar path for layer 1 (Cin=2).
// EXEC is all-ones at every WMMA (full 256-thread blocks, no divergence).
// ---------------------------------------------------------------------------
#define BM 256
#define BN 64
#define BK 32

__global__ __launch_bounds__(256) void svd_conv_wmma(
    const unsigned short* __restrict__ act,   // [B][H][W][Cin] bf16 (NHWC)
    const unsigned short* __restrict__ wpk,   // [Cout2][Kpad]  bf16
    const float* __restrict__ bias,           // [Cout2]
    float* __restrict__ out,                  // [B][Cout2][H][W] fp32 (NCHW)
    int Cin, int cinShift, int Cout2, int Kpad) {
  __shared__ __attribute__((aligned(16))) unsigned short As[2][BM * BK];
  __shared__ __attribute__((aligned(16))) unsigned short Bs[2][BN * BK];

  const int tid   = threadIdx.x;
  const int lane  = tid & 31;
  const int wid   = tid >> 5;
  const int waveM = wid & 3;                  // 0..3 -> 64 rows each
  const int waveN = wid >> 2;                 // 0..1 -> 32 cols each

  const int K     = Cin * 9;
  const int pbase = blockIdx.x * BM;          // pixel base (one image, 4 rows)
  const int b     = pbase >> 12;
  const int yx0   = pbase & (HW - 1);
  const int ncol0 = blockIdx.y * BN;

  // ---- tile loaders (double buffered) ------------------------------------
  auto loadA = [&](int k0, int buf) {
    if (cinShift >= 5) {
      // vector path: 256 rows x 4 chunks of 8 bf16 (16B) = 1024 chunks
      for (int t = 0; t < 4; ++t) {
        int idx = tid + (t << 8);             // 0..1023
        int i = idx >> 2, ch = idx & 3;
        int kk0 = ch << 3;                    // 0,8,16,24
        int k = k0 + kk0;
        uint4 val = {0u, 0u, 0u, 0u};
        if (k < K) {
          int c0  = k & (Cin - 1);
          int dxy = k >> cinShift;            // one 3x3 tap for whole chunk
          int dy  = dxy / 3;
          int dx  = dxy - dy * 3;
          int yx  = yx0 + i;
          int y   = (yx >> 6) + dy - 1;
          int x   = (yx & 63) + dx - 1;
          if ((unsigned)y < 64u && (unsigned)x < 64u)
            val = *(const uint4*)&act[(((size_t)b << 12) + (y << 6) + x) * Cin + c0];
        }
        *(uint4*)&As[buf][i * BK + kk0] = val;
      }
    } else {
      // scalar path (layer 1, Cin=2): 8192 elems, 32 per thread
      for (int t = 0; t < 32; ++t) {
        int idx = tid + (t << 8);
        int i = idx >> 5, kk = idx & 31;
        int k = k0 + kk;
        unsigned short v = 0;
        if (k < K) {
          int c   = k & (Cin - 1);
          int dxy = k >> cinShift;
          int dy  = dxy / 3;
          int dx  = dxy - dy * 3;
          int yx  = yx0 + i;
          int y   = (yx >> 6) + dy - 1;
          int x   = (yx & 63) + dx - 1;
          if ((unsigned)y < 64u && (unsigned)x < 64u)
            v = act[(((size_t)b << 12) + (y << 6) + x) * Cin + c];
        }
        As[buf][i * BK + kk] = v;
      }
    }
  };
  auto loadB = [&](int k0, int buf) {
    // 64 x 32 bf16 = 256 chunks of 16B, one per thread
    int n = tid >> 2, kk0 = (tid & 3) << 3;
    uint4 val = *(const uint4*)&wpk[(size_t)(ncol0 + n) * Kpad + k0 + kk0];
    *(uint4*)&Bs[buf][n * BK + kk0] = val;
  };

  v8f acc[4][2];
  for (int mi = 0; mi < 4; ++mi)
    for (int ni = 0; ni < 2; ++ni)
      for (int r = 0; r < 8; ++r) acc[mi][ni][r] = 0.f;

  const int nsteps = Kpad / BK;
  loadA(0, 0);
  loadB(0, 0);
  int cur = 0;

  const int mrow   = lane & 15;
  const int kbaseA = (lane < 16) ? 0 : 16;
  const int kbaseB = (lane < 16) ? 0 : 16;

  for (int s = 0; s < nsteps; ++s) {
    __syncthreads();
    if (s + 1 < nsteps) {                     // stage next tile in alt buffer
      loadA((s + 1) * BK, cur ^ 1);
      loadB((s + 1) * BK, cur ^ 1);
    }
    if (s + 2 < nsteps)                       // gfx1250 global_prefetch_b8
      __builtin_prefetch(&wpk[(size_t)(ncol0 + (tid >> 2)) * Kpad + (s + 2) * BK], 0, 0);

    // ---- fragments: lane half h holds k = 16h..16h+15 (same order for A and B)
    const unsigned int* As32 = (const unsigned int*)As[cur];
    const unsigned int* Bs32 = (const unsigned int*)Bs[cur];

    FragBF bfrag[2];
    for (int ni = 0; ni < 2; ++ni) {
      int n = waveN * 32 + ni * 16 + mrow;
      for (int vi = 0; vi < 8; ++vi) {
        int k2 = kbaseB + (vi << 1);
        bfrag[ni].d[vi] = Bs32[(n * BK + k2) >> 1];
      }
    }
    for (int mi = 0; mi < 4; ++mi) {
      FragBF afrag;
      int m = waveM * 64 + mi * 16 + mrow;
      for (int vi = 0; vi < 8; ++vi) {
        int k2 = kbaseA + (vi << 1);
        afrag.d[vi] = As32[(m * BK + k2) >> 1];
      }
      for (int ni = 0; ni < 2; ++ni)
        acc[mi][ni] = __builtin_amdgcn_wmma_f32_16x16x32_bf16(
            false, afrag.v, false, bfrag[ni].v,
            (short)0, acc[mi][ni], false, false);
    }
    cur ^= 1;
  }

  // ---- epilogue: C/D layout -> (m,n); add combined bias
  const int nlocal = lane & 15;
  const int moff   = (lane >> 4) << 3;        // 0 or 8
  for (int mi = 0; mi < 4; ++mi)
    for (int ni = 0; ni < 2; ++ni) {
      int n = ncol0 + waveN * 32 + ni * 16 + nlocal;
      float bv = bias[n];
      for (int r = 0; r < 8; ++r) {
        int i = waveM * 64 + mi * 16 + r + moff;
        out[(((size_t)b * Cout2 + n) << 12) + yx0 + i] = acc[mi][ni][r] + bv;
      }
    }
}

// ---------------------------------------------------------------------------
// BatchNorm: per-(b,c) plane partial sums -> float atomics into stats[2C]
// ---------------------------------------------------------------------------
__global__ __launch_bounds__(256) void svd_bn_stats(const float* __restrict__ x,
                                                    float* __restrict__ stats,
                                                    int C) {
  int c = blockIdx.x % C;
  int b = blockIdx.x / C;
  const float* p = x + (((size_t)b * C + c) << 12);
  float s = 0.f, s2 = 0.f;
  for (int i = threadIdx.x; i < HW; i += 256) {
    float v = p[i];
    s += v; s2 += v * v;
  }
  __shared__ float rs[256], rq[256];
  rs[threadIdx.x] = s; rq[threadIdx.x] = s2;
  __syncthreads();
  for (int off = 128; off > 0; off >>= 1) {
    if (threadIdx.x < off) {
      rs[threadIdx.x] += rs[threadIdx.x + off];
      rq[threadIdx.x] += rq[threadIdx.x + off];
    }
    __syncthreads();
  }
  if (threadIdx.x == 0) {
    atomicAdd(&stats[2 * c + 0], rs[0]);
    atomicAdd(&stats[2 * c + 1], rq[0]);
  }
}

// Reads NCHW fp32 conv output; optional fp32 write-back (in place) and
// optional bf16 NHWC write for the next conv layer.
__global__ void svd_bn_apply(const float* __restrict__ xin,
                             float* __restrict__ xoutf,
                             unsigned short* __restrict__ xoutb,
                             const float* __restrict__ stats,
                             const float* __restrict__ g,
                             const float* __restrict__ be,
                             int C, int writeF32, int writeBf16) {
  size_t total = (size_t)BATCH * C * HW;
  size_t idx = (size_t)blockIdx.x * blockDim.x + threadIdx.x;
  if (idx >= total) return;
  size_t plane = idx >> 12;
  int yx = (int)(idx & (HW - 1));
  int c = (int)(plane % C);
  int b = (int)(plane / C);
  const float inv = 1.f / (float)NPIX;
  float mean = stats[2 * c + 0] * inv;
  float var  = stats[2 * c + 1] * inv - mean * mean;
  float y = g[c] * (xin[idx] - mean) * rsqrtf(var + 1e-5f) + be[c];
  y = fmaxf(y, 0.f);
  if (writeF32)  xoutf[idx] = y;
  if (writeBf16) xoutb[(((size_t)b << 12) + yx) * C + c] = f32_to_bf16(y);
}

// ---------------------------------------------------------------------------
// Poolings over fp32 h [B][256][H][W]
// ---------------------------------------------------------------------------
__global__ void svd_pool_u(const float* __restrict__ h, float* __restrict__ pu) {
  int idx = blockIdx.x * blockDim.x + threadIdx.x;   // B*256*64
  if (idx >= BATCH * 256 * 64) return;
  const float* p = h + ((size_t)idx << 6);           // row of 64
  float s = 0.f;
  for (int x = 0; x < 64; ++x) s += p[x];
  pu[idx] = s * (1.f / 64.f);
}
__global__ void svd_pool_v(const float* __restrict__ h, float* __restrict__ pv) {
  int idx = blockIdx.x * blockDim.x + threadIdx.x;   // B*256*64 (x fastest)
  if (idx >= BATCH * 256 * 64) return;
  int x  = idx & 63;
  int bc = idx >> 6;
  const float* p = h + (((size_t)bc) << 12) + x;
  float s = 0.f;
  for (int y = 0; y < 64; ++y) s += p[y << 6];
  pv[idx] = s * (1.f / 64.f);
}
__global__ void svd_pool_s(const float* __restrict__ pu, float* __restrict__ ps) {
  int idx = blockIdx.x * blockDim.x + threadIdx.x;   // B*256
  if (idx >= BATCH * 256) return;
  const float* p = pu + ((size_t)idx << 6);
  float s = 0.f;
  for (int y = 0; y < 64; ++y) s += p[y];
  ps[idx] = s * (1.f / 64.f);
}

// out[b][o][m] = relu(sum_c P[b][c][m] * W[o][c] + bias[o]);  C fixed = 256
__global__ void svd_head(const float* __restrict__ P, const float* __restrict__ W,
                         const float* __restrict__ bias, float* __restrict__ out,
                         int O, int Mdim) {
  int idx = blockIdx.x * blockDim.x + threadIdx.x;
  int total = BATCH * O * Mdim;
  if (idx >= total) return;
  int m = idx % Mdim;
  int o = (idx / Mdim) % O;
  int b = idx / (Mdim * O);
  const float* p = P + (size_t)b * 256 * Mdim + m;
  float acc = bias[o];
  for (int c = 0; c < 256; ++c) acc += p[(size_t)c * Mdim] * W[o * 256 + c];
  out[idx] = fmaxf(acc, 0.f);
}

// ---------------------------------------------------------------------------
// Gram-Schmidt: head [B][16][Mdim=64] -> out [B][64][8][2]; one block per b.
// ---------------------------------------------------------------------------
__device__ __forceinline__ float svd_reduce64(float v, volatile float* red) {
  int m = threadIdx.x;
  red[m] = v;
  __syncthreads();
  for (int off = 32; off > 0; off >>= 1) {
    if (m < off) red[m] += red[m + off];
    __syncthreads();
  }
  float r = red[0];
  __syncthreads();
  return r;
}

__global__ __launch_bounds__(64) void svd_gram(const float* __restrict__ head,
                                               float* __restrict__ out) {
  int b = blockIdx.x;
  int m = threadIdx.x;   // 0..63
  __shared__ float br_[RANK][64], bi_[RANK][64];
  __shared__ float red[64];

  for (int j = 0; j < RANK; ++j) {
    float vr = head[((size_t)b * 16 + 2 * j + 0) * 64 + m];
    float vi = head[((size_t)b * 16 + 2 * j + 1) * 64 + m];
    for (int t = 0; t < j; ++t) {
      float dot = svd_reduce64(vr * br_[t][m] + vi * bi_[t][m], red);
      float den = svd_reduce64(br_[t][m] * br_[t][m] + bi_[t][m] * bi_[t][m], red) + 1e-8f;
      float coef = dot / den;
      vr -= coef * br_[t][m];
      vi -= coef * bi_[t][m];
    }
    float nrm = sqrtf(svd_reduce64(vr * vr + vi * vi, red) + 1e-8f);
    vr /= nrm; vi /= nrm;
    br_[j][m] = vr; bi_[j][m] = vi;
    __syncthreads();
    out[(((size_t)b * 64 + m) * RANK + j) * 2 + 0] = vr;
    out[(((size_t)b * 64 + m) * RANK + j) * 2 + 1] = vi;
  }
}

// ---------------------------------------------------------------------------
// Host orchestration
// ---------------------------------------------------------------------------
extern "C" void kernel_launch(void* const* d_in, const int* in_sizes, int n_in,
                              void* d_out, int out_size, void* d_ws, size_t ws_size,
                              hipStream_t stream) {
  (void)in_sizes; (void)n_in; (void)out_size; (void)ws_size;
  const float* x   = (const float*)d_in[0];
  const float* w1r = (const float*)d_in[1];  const float* b1r = (const float*)d_in[2];
  const float* w1i = (const float*)d_in[3];  const float* b1i = (const float*)d_in[4];
  const float* g1  = (const float*)d_in[5];  const float* be1 = (const float*)d_in[6];
  const float* w2r = (const float*)d_in[7];  const float* b2r = (const float*)d_in[8];
  const float* w2i = (const float*)d_in[9];  const float* b2i = (const float*)d_in[10];
  const float* g2  = (const float*)d_in[11]; const float* be2 = (const float*)d_in[12];
  const float* w3r = (const float*)d_in[13]; const float* b3r = (const float*)d_in[14];
  const float* w3i = (const float*)d_in[15]; const float* b3i = (const float*)d_in[16];
  const float* g3  = (const float*)d_in[17]; const float* be3 = (const float*)d_in[18];
  const float* wu  = (const float*)d_in[19]; const float* bu  = (const float*)d_in[20];
  const float* wsm = (const float*)d_in[21]; const float* bsv = (const float*)d_in[22];
  const float* wv  = (const float*)d_in[23]; const float* bv  = (const float*)d_in[24];

  // ---- workspace carve-out
  char* base = (char*)d_ws;
  size_t off = 0;
  auto carve = [&](size_t bytes) -> char* {
    char* p = base + off;
    off = (off + bytes + 255) & ~(size_t)255;
    return p;
  };
  const int Kp1 = 32, Kp2 = 576, Kp3 = 1152;
  unsigned short* wpk1 = (unsigned short*)carve((size_t)64  * Kp1 * 2);
  unsigned short* wpk2 = (unsigned short*)carve((size_t)128 * Kp2 * 2);
  unsigned short* wpk3 = (unsigned short*)carve((size_t)256 * Kp3 * 2);
  float* bias1 = (float*)carve(64 * 4);
  float* bias2 = (float*)carve(128 * 4);
  float* bias3 = (float*)carve(256 * 4);
  float* stats = (float*)carve(2 * 256 * 4);
  unsigned short* act0 = (unsigned short*)carve((size_t)BATCH * 128 * HW * 2);
  unsigned short* act1 = (unsigned short*)carve((size_t)BATCH * 128 * HW * 2);
  float* convf = (float*)carve((size_t)BATCH * 256 * HW * 4);
  float* pu    = (float*)carve((size_t)BATCH * 256 * 64 * 4);
  float* pv    = (float*)carve((size_t)BATCH * 256 * 64 * 4);
  float* ps    = (float*)carve((size_t)BATCH * 256 * 4);
  float* uh    = (float*)carve((size_t)BATCH * 16 * 64 * 4);
  float* vh    = (float*)carve((size_t)BATCH * 16 * 64 * 4);

  float* out_u = (float*)d_out;                       // 128*64*8*2 = 131072
  float* out_s = out_u + 131072;                      // 128*8      = 1024
  float* out_v = out_s + 1024;                        // 131072

  const int numM = NPIX / BM;                         // 2048

  // ---- pack weights / biases / input
  svd_pack_weights<<<(2*64*Kp1  + 255)/256, 256, 0, stream>>>(w1r, w1i, wpk1, 2,   32,  Kp1);
  svd_pack_weights<<<(2*128*Kp2 + 255)/256, 256, 0, stream>>>(w2r, w2i, wpk2, 64,  64,  Kp2);
  svd_pack_weights<<<(2*256*Kp3 + 255)/256, 256, 0, stream>>>(w3r, w3i, wpk3, 128, 128, Kp3);
  svd_pack_bias<<<1, 256, 0, stream>>>(b1r, b1i, bias1, 32);
  svd_pack_bias<<<1, 256, 0, stream>>>(b2r, b2i, bias2, 64);
  svd_pack_bias<<<2, 256, 0, stream>>>(b3r, b3i, bias3, 128);
  svd_pack_input<<<(BATCH*HW*2 + 255)/256, 256, 0, stream>>>(x, act0);

  // ---- layer 1: 2 -> 64 (scalar A path)
  svd_conv_wmma<<<dim3(numM, 1), 256, 0, stream>>>(act0, wpk1, bias1, convf, 2, 1, 64, Kp1);
  hipMemsetAsync(stats, 0, 2 * 64 * 4, stream);
  svd_bn_stats<<<64 * BATCH, 256, 0, stream>>>(convf, stats, 64);
  {
    size_t total = (size_t)BATCH * 64 * HW;
    svd_bn_apply<<<(unsigned)((total + 255)/256), 256, 0, stream>>>(
        convf, convf, act1, stats, g1, be1, 64, 0, 1);
  }

  // ---- layer 2: 64 -> 128 (vector A path)
  svd_conv_wmma<<<dim3(numM, 2), 256, 0, stream>>>(act1, wpk2, bias2, convf, 64, 6, 128, Kp2);
  hipMemsetAsync(stats, 0, 2 * 128 * 4, stream);
  svd_bn_stats<<<128 * BATCH, 256, 0, stream>>>(convf, stats, 128);
  {
    size_t total = (size_t)BATCH * 128 * HW;
    svd_bn_apply<<<(unsigned)((total + 255)/256), 256, 0, stream>>>(
        convf, convf, act0, stats, g2, be2, 128, 0, 1);
  }

  // ---- layer 3: 128 -> 256 (keep fp32 h in place for poolings)
  svd_conv_wmma<<<dim3(numM, 4), 256, 0, stream>>>(act0, wpk3, bias3, convf, 128, 7, 256, Kp3);
  hipMemsetAsync(stats, 0, 2 * 256 * 4, stream);
  svd_bn_stats<<<256 * BATCH, 256, 0, stream>>>(convf, stats, 256);
  {
    size_t total = (size_t)BATCH * 256 * HW;
    svd_bn_apply<<<(unsigned)((total + 255)/256), 256, 0, stream>>>(
        convf, convf, act1, stats, g3, be3, 256, 1, 0);
  }

  // ---- poolings
  svd_pool_u<<<(BATCH*256*64 + 255)/256, 256, 0, stream>>>(convf, pu);
  svd_pool_v<<<(BATCH*256*64 + 255)/256, 256, 0, stream>>>(convf, pv);
  svd_pool_s<<<(BATCH*256 + 255)/256, 256, 0, stream>>>(pu, ps);

  // ---- heads
  svd_head<<<(BATCH*16*64 + 255)/256, 256, 0, stream>>>(pu, wu,  bu,  uh,    16, 64);
  svd_head<<<(BATCH*8     + 255)/256, 256, 0, stream>>>(ps, wsm, bsv, out_s,  8,  1);
  svd_head<<<(BATCH*16*64 + 255)/256, 256, 0, stream>>>(pv, wv,  bv,  vh,    16, 64);

  // ---- Gram-Schmidt to final u, v
  svd_gram<<<BATCH, 64, 0, stream>>>(uh, out_u);
  svd_gram<<<BATCH, 64, 0, stream>>>(vh, out_v);
}
